// InteractionLayer_49478023250265
// MI455X (gfx1250) — matrix operand order repaired and checked by reading, not compiled, not run on hardware
//
#include <hip/hip_runtime.h>
#include <hip/hip_bf16.h>

// ---------------------------------------------------------------------------
// PhysNet interaction layer for MI455X (gfx1250, wave32, WMMA bf16).
// Fused pipeline: atom features -> edge filter GEMM + gather/segment-sum ->
// residual stack + output. All GEMMs via v_wmma_f32_16x16x32_bf16.
// ---------------------------------------------------------------------------

typedef __attribute__((ext_vector_type(16))) __bf16 v16bf;
typedef __attribute__((ext_vector_type(4)))  __bf16 v4bf;
typedef __attribute__((ext_vector_type(8)))  float  v8f;

#define LN2F 0.6931471805599453f

__device__ __forceinline__ float ssp(float x) {
    // shifted softplus: ln(1+e^x) - ln2, numerically stable
    float ax = __builtin_fabsf(x);
    return fmaxf(x, 0.0f) + __logf(1.0f + __expf(-ax)) - LN2F;
}

__device__ __forceinline__ v8f wmma_bf16(v16bf a, v16bf b, v8f c) {
    return __builtin_amdgcn_wmma_f32_16x16x32_bf16(
        /*neg_a=*/false, a, /*neg_b=*/false, b,
        /*c_mod=*/(short)0, c, /*reuse_a=*/false, /*reuse_b=*/false);
}

// ---------------------------------------------------------------------------
// B-operand staging: W is row-major [K][128] f32 in global. WMMA B 32x16
// (16-bit) layout: lane = (n%16) + 16*((klocal%16)>>3), h = (klocal&7) + 8*(klocal>>4).
// LDS layout: [ntile(8)][kk(K/32)][lane(32)][h(16)] bf16.  blockDim.x == 128,
// thread tid owns column n = tid.  4 consecutive k pack into one b64 store.
// ---------------------------------------------------------------------------
__device__ __forceinline__ void stage_weight_B(const float* __restrict__ W,
                                               __bf16* wlds, int K, int tid) {
    int n = tid;
    int ntile = n >> 4;
    int nlo = n & 15;
    int nkk = K >> 5;
#pragma unroll 4
    for (int k0 = 0; k0 < K; k0 += 4) {
        float f0 = W[(k0 + 0) * 128 + n];
        float f1 = W[(k0 + 1) * 128 + n];
        float f2 = W[(k0 + 2) * 128 + n];
        float f3 = W[(k0 + 3) * 128 + n];
        int kk = k0 >> 5;
        int klocal = k0 & 31;
        int h = (klocal & 7) + ((klocal >> 4) << 3); // h, h+1, h+2, h+3
        int lane = nlo + (((klocal & 15) >> 3) << 4);
        v4bf pk;
        pk.x = (__bf16)f0; pk.y = (__bf16)f1; pk.z = (__bf16)f2; pk.w = (__bf16)f3;
        *(v4bf*)(wlds + (((ntile * nkk + kk) * 32 + lane) * 16 + h)) = pk;
    }
}

// ---------------------------------------------------------------------------
// A-operand staging from global row-major [rows][ld] f32. WMMA A 16x32
// (16-bit) layout: lane = m + 16*((klocal%16)>>3), h = (klocal&7) + 8*(klocal>>4).
// LDS layout (per wave): [kk(K/32)][lane(32)][h(16)] bf16.
// Lane m = lane&15, covers K/2 contiguous k starting at (lane>>4)*(K/2).
// ---------------------------------------------------------------------------
__device__ __forceinline__ void stage_A_tile(const float* __restrict__ src, int ld,
                                             int row0, int maxrow, __bf16* aw,
                                             int K, int lane, bool do_ssp) {
    int m = lane & 15;
    int row = row0 + m; if (row > maxrow) row = maxrow;
    const float* sp = src + (long)row * ld;
    int khalf = K >> 1;
    int kstart = (lane >> 4) * khalf;
#pragma unroll 4
    for (int k0 = kstart; k0 < kstart + khalf; k0 += 4) {
        float4 v = *(const float4*)(sp + k0);
        float f0 = v.x, f1 = v.y, f2 = v.z, f3 = v.w;
        if (do_ssp) { f0 = ssp(f0); f1 = ssp(f1); f2 = ssp(f2); f3 = ssp(f3); }
        int kk = k0 >> 5;
        int klocal = k0 & 31;
        int h = (klocal & 7) + ((klocal >> 4) << 3); // h..h+3 contiguous
        int lane_a = m + (((klocal & 15) >> 3) << 4);
        v4bf pk;
        pk.x = (__bf16)f0; pk.y = (__bf16)f1; pk.z = (__bf16)f2; pk.w = (__bf16)f3;
        *(v4bf*)(aw + ((kk * 32 + lane_a) * 16 + h)) = pk;
    }
}

// ---------------------------------------------------------------------------
// Re-stage a D-layout register tile (8 ntiles x 8 rows per lane) as the next
// GEMM's A operand.  D: lane holds (M = r + 8*(lane>>4), N = nt*16 + lane&15).
// ---------------------------------------------------------------------------
__device__ __forceinline__ void stage_D_to_A(const float (&v)[8][8], __bf16* aw,
                                             int lane, bool do_ssp) {
    int col0 = lane & 15;
    int mloc = (lane >> 4) << 3;
#pragma unroll
    for (int nt = 0; nt < 8; ++nt) {
        int n = nt * 16 + col0;        // this is the k index of the next GEMM
        int kk = n >> 5;
        int klocal = n & 31;
        int h = (klocal & 7) + ((klocal >> 4) << 3);
        int lgrp = (((klocal & 15) >> 3) << 4);
#pragma unroll
        for (int r = 0; r < 8; ++r) {
            float f = v[nt][r];
            if (do_ssp) f = ssp(f);
            aw[((kk * 32 + (mloc + r + lgrp)) * 16 + h)] = (__bf16)f;
        }
    }
}

// ---------------------------------------------------------------------------
// Kernel 0: zero the scatter accumulator.
// ---------------------------------------------------------------------------
__global__ void physnet_zero_kernel(float4* __restrict__ p, int n4) {
    int i = blockIdx.x * 256 + threadIdx.x;
    if (i < n4) p[i] = make_float4(0.f, 0.f, 0.f, 0.f);
}

// ---------------------------------------------------------------------------
// Kernel 1: xa = ssp(x); xi = ssp(xa@W_i + b_i); xjf = ssp(xa@W_j + b_j)
// Block = 128 threads = 4 waves; 64 atoms per block (M=16 per wave).
// ---------------------------------------------------------------------------
__global__ __launch_bounds__(128) void physnet_atom_feat_kernel(
    const float* __restrict__ x,
    const float* __restrict__ W_i, const float* __restrict__ b_i,
    const float* __restrict__ W_j, const float* __restrict__ b_j,
    float* __restrict__ xi, float* __restrict__ xjf, int n_atoms) {
    __shared__ __align__(32) __bf16 wlds[8 * 4 * 32 * 16];   // 32 KB (one matrix)
    __shared__ __align__(32) __bf16 alds[4][4 * 32 * 16];    // 16 KB (A tiles)
    __shared__ float bias_s[128];

    int tid = threadIdx.x, wave = tid >> 5, lane = tid & 31;
    int atom0 = blockIdx.x * 64 + wave * 16;
    bool full = (atom0 + 16 <= n_atoms);   // wave-uniform: no store guards needed

    stage_A_tile(x, 128, atom0, n_atoms - 1, alds[wave], 128, lane, /*ssp=*/true);
    __syncthreads();

    // A fragments are shared by both weight matrices and all 8 N-tiles.
    v16bf afrag[4];
#pragma unroll
    for (int kk = 0; kk < 4; ++kk)
        afrag[kk] = *(const v16bf*)(alds[wave] + (kk * 32 + lane) * 16);

    for (int mat = 0; mat < 2; ++mat) {
        __syncthreads();  // protect wlds/bias from previous iteration's readers
        stage_weight_B(mat ? W_j : W_i, wlds, 128, tid);
        bias_s[tid] = (mat ? b_j : b_i)[tid];
        __syncthreads();

        float* outp = mat ? xjf : xi;
#pragma unroll
        for (int nt = 0; nt < 8; ++nt) {
            v8f c = {0.f, 0.f, 0.f, 0.f, 0.f, 0.f, 0.f, 0.f};
#pragma unroll
            for (int kk = 0; kk < 4; ++kk) {
                v16bf b = *(const v16bf*)(wlds + ((nt * 4 + kk) * 32 + lane) * 16);
                c = wmma_bf16(afrag[kk], b, c);
            }
            int col = nt * 16 + (lane & 15);
            float bv = bias_s[col];
            int mb = (lane >> 4) << 3;
            float* po = outp + (long)(atom0 + mb) * 128 + col;
            if (full) {
#pragma unroll
                for (int r = 0; r < 8; ++r) po[r * 128] = ssp(c[r] + bv);
            } else {
#pragma unroll
                for (int r = 0; r < 8; ++r)
                    if (atom0 + mb + r < n_atoms) po[r * 128] = ssp(c[r] + bv);
            }
        }
    }
}

// ---------------------------------------------------------------------------
// Kernel 2: g = rbf @ W_k2f (WMMA, K=64); pxj = g * xjf[idx_j];
//           xj = segment_sum(pxj, idx_i)  (sorted idx_i -> run-length + atomics)
// Block = 128 threads = 4 waves; 64 edges per block; thread t owns feature t.
// ---------------------------------------------------------------------------
__global__ __launch_bounds__(128) void physnet_edge_kernel(
    const float* __restrict__ rbf, const float* __restrict__ W_k2f,
    const int* __restrict__ idx_i, const int* __restrict__ idx_j,
    const float* __restrict__ xjf, float* __restrict__ xj, int n_edges) {
    __shared__ __align__(32) __bf16 wlds[8 * 2 * 32 * 16];   // 16 KB
    __shared__ __align__(32) __bf16 alds[4][2 * 32 * 16];    //  8 KB
    __shared__ float glds[64 * 128];                         // 32 KB
    __shared__ int ii_s[64], ij_s[64];

    int tid = threadIdx.x, wave = tid >> 5, lane = tid & 31;
    int e0 = blockIdx.x * 64;

    stage_weight_B(W_k2f, wlds, 64, tid);
    if (tid < 64) {
        int e = e0 + tid; if (e > n_edges - 1) e = n_edges - 1;
        ii_s[tid] = idx_i[e];
        ij_s[tid] = idx_j[e];
    }
    stage_A_tile(rbf, 64, e0 + wave * 16, n_edges - 1, alds[wave], 64, lane, false);
    __syncthreads();

    v16bf afrag[2];
#pragma unroll
    for (int kk = 0; kk < 2; ++kk)
        afrag[kk] = *(const v16bf*)(alds[wave] + (kk * 32 + lane) * 16);

#pragma unroll
    for (int nt = 0; nt < 8; ++nt) {
        v8f c = {0.f, 0.f, 0.f, 0.f, 0.f, 0.f, 0.f, 0.f};
#pragma unroll
        for (int kk = 0; kk < 2; ++kk) {
            v16bf b = *(const v16bf*)(wlds + ((nt * 2 + kk) * 32 + lane) * 16);
            c = wmma_bf16(afrag[kk], b, c);
        }
        int col = nt * 16 + (lane & 15);
        int mb = (lane >> 4) << 3;
#pragma unroll
        for (int r = 0; r < 8; ++r)
            glds[(wave * 16 + mb + r) * 128 + col] = c[r];
    }
    __syncthreads();

    // Scatter phase: feature column per thread; run-length over sorted idx_i.
    int f = tid;
    int nvalid = n_edges - e0; if (nvalid > 64) nvalid = 64;
    float acc = 0.f;
    int cur = ii_s[0];
    for (int e = 0; e < nvalid; ++e) {
        if (e + 4 < nvalid)
            __builtin_prefetch(&xjf[(long)ij_s[e + 4] * 128 + f], 0, 1);
        int seg = ii_s[e];
        float v = glds[e * 128 + f] * xjf[(long)ij_s[e] * 128 + f];
        if (seg != cur) {                // block-uniform branch (same e for all)
            atomicAdd(&xj[(long)cur * 128 + f], acc);
            acc = 0.f;
            cur = seg;
        }
        acc += v;
    }
    atomicAdd(&xj[(long)cur * 128 + f], acc);
}

// ---------------------------------------------------------------------------
// Kernel 3: m = xi + xj; 3x residual (two WMMA GEMMs each, C seeded with m);
//           out = u*x + ssp(m)@W_out + b_out.  m lives in registers (D layout).
// ---------------------------------------------------------------------------
__global__ __launch_bounds__(128) void physnet_residual_out_kernel(
    const float* __restrict__ xi, const float* __restrict__ xj,
    const float* __restrict__ x,
    const float* __restrict__ W_res1, const float* __restrict__ b_res1,
    const float* __restrict__ W_res2, const float* __restrict__ b_res2,
    const float* __restrict__ W_out, const float* __restrict__ b_out,
    const float* __restrict__ u, float* __restrict__ out, int n_atoms) {
    __shared__ __align__(32) __bf16 wlds[8 * 4 * 32 * 16];   // 32 KB
    __shared__ __align__(32) __bf16 alds[4][4 * 32 * 16];    // 16 KB
    __shared__ float bias_s[128];
    __shared__ float u_s[128];

    int tid = threadIdx.x, wave = tid >> 5, lane = tid & 31;
    int atom0 = blockIdx.x * 64 + wave * 16;
    int col0 = lane & 15, mb = (lane >> 4) << 3;
    bool full = (atom0 + 16 <= n_atoms);   // wave-uniform

    float mreg[8][8];
#pragma unroll
    for (int nt = 0; nt < 8; ++nt) {
        int col = nt * 16 + col0;
#pragma unroll
        for (int r = 0; r < 8; ++r) {
            int a = atom0 + mb + r; if (a >= n_atoms) a = n_atoms - 1;
            long o = (long)a * 128 + col;
            mreg[nt][r] = xi[o] + xj[o];
        }
    }

    for (int l = 0; l < 3; ++l) {
        float treg[8][8];
        // ---- GEMM 1: t = ssp( ssp(m) @ W1 + b1 ) ----
        __syncthreads();
        stage_D_to_A(mreg, alds[wave], lane, /*ssp=*/true);
        stage_weight_B(W_res1 + l * 128 * 128, wlds, 128, tid);
        bias_s[tid] = b_res1[l * 128 + tid];
        __syncthreads();
        {
            v16bf afrag[4];
#pragma unroll
            for (int kk = 0; kk < 4; ++kk)
                afrag[kk] = *(const v16bf*)(alds[wave] + (kk * 32 + lane) * 16);
#pragma unroll
            for (int nt = 0; nt < 8; ++nt) {
                v8f c = {0.f, 0.f, 0.f, 0.f, 0.f, 0.f, 0.f, 0.f};
#pragma unroll
                for (int kk = 0; kk < 4; ++kk) {
                    v16bf b = *(const v16bf*)(wlds + ((nt * 4 + kk) * 32 + lane) * 16);
                    c = wmma_bf16(afrag[kk], b, c);
                }
                float bv = bias_s[nt * 16 + col0];
#pragma unroll
                for (int r = 0; r < 8; ++r) treg[nt][r] = ssp(c[r] + bv);
            }
        }
        // ---- GEMM 2: m = m + t @ W2 + b2  (C seeded with m) ----
        __syncthreads();
        stage_D_to_A(treg, alds[wave], lane, /*ssp=*/false);
        stage_weight_B(W_res2 + l * 128 * 128, wlds, 128, tid);
        bias_s[tid] = b_res2[l * 128 + tid];
        __syncthreads();
        {
            v16bf afrag[4];
#pragma unroll
            for (int kk = 0; kk < 4; ++kk)
                afrag[kk] = *(const v16bf*)(alds[wave] + (kk * 32 + lane) * 16);
#pragma unroll
            for (int nt = 0; nt < 8; ++nt) {
                v8f c;
#pragma unroll
                for (int r = 0; r < 8; ++r) c[r] = mreg[nt][r];
#pragma unroll
                for (int kk = 0; kk < 4; ++kk) {
                    v16bf b = *(const v16bf*)(wlds + ((nt * 4 + kk) * 32 + lane) * 16);
                    c = wmma_bf16(afrag[kk], b, c);
                }
                float bv = bias_s[nt * 16 + col0];
#pragma unroll
                for (int r = 0; r < 8; ++r) mreg[nt][r] = c[r] + bv;
            }
        }
    }

    // ---- output: out = u*x + ssp(m) @ W_out + b_out ----
    __syncthreads();
    stage_D_to_A(mreg, alds[wave], lane, /*ssp=*/true);
    stage_weight_B(W_out, wlds, 128, tid);
    bias_s[tid] = b_out[tid];
    u_s[tid] = u[tid];
    __syncthreads();
    {
        v16bf afrag[4];
#pragma unroll
        for (int kk = 0; kk < 4; ++kk)
            afrag[kk] = *(const v16bf*)(alds[wave] + (kk * 32 + lane) * 16);
#pragma unroll
        for (int nt = 0; nt < 8; ++nt) {
            v8f c = {0.f, 0.f, 0.f, 0.f, 0.f, 0.f, 0.f, 0.f};
#pragma unroll
            for (int kk = 0; kk < 4; ++kk) {
                v16bf b = *(const v16bf*)(wlds + ((nt * 4 + kk) * 32 + lane) * 16);
                c = wmma_bf16(afrag[kk], b, c);
            }
            int col = nt * 16 + col0;
            float bv = bias_s[col], uv = u_s[col];
            const float* px = x + (long)(atom0 + mb) * 128 + col;
            float* po = out + (long)(atom0 + mb) * 128 + col;
            if (full) {
#pragma unroll
                for (int r = 0; r < 8; ++r) po[r * 128] = uv * px[r * 128] + c[r] + bv;
            } else {
#pragma unroll
                for (int r = 0; r < 8; ++r)
                    if (atom0 + mb + r < n_atoms)
                        po[r * 128] = uv * px[r * 128] + c[r] + bv;
            }
        }
    }
}

// ---------------------------------------------------------------------------
extern "C" void kernel_launch(void* const* d_in, const int* in_sizes, int n_in,
                              void* d_out, int out_size, void* d_ws, size_t ws_size,
                              hipStream_t stream) {
    const float* x      = (const float*)d_in[0];
    const float* rbf    = (const float*)d_in[1];
    const int*   idx_i  = (const int*)d_in[2];
    const int*   idx_j  = (const int*)d_in[3];
    const float* W_k2f  = (const float*)d_in[4];
    const float* W_i    = (const float*)d_in[5];
    const float* b_i    = (const float*)d_in[6];
    const float* W_j    = (const float*)d_in[7];
    const float* b_j    = (const float*)d_in[8];
    const float* W_res1 = (const float*)d_in[9];
    const float* b_res1 = (const float*)d_in[10];
    const float* W_res2 = (const float*)d_in[11];
    const float* b_res2 = (const float*)d_in[12];
    const float* W_out  = (const float*)d_in[13];
    const float* b_out  = (const float*)d_in[14];
    const float* u      = (const float*)d_in[15];
    float* out = (float*)d_out;

    int NA = in_sizes[0] / 128;
    int NE = in_sizes[2];

    float* xi  = (float*)d_ws;
    float* xjf = xi  + (size_t)NA * 128;
    float* xj  = xjf + (size_t)NA * 128;

    int n4 = (NA * 128) / 4;
    physnet_zero_kernel<<<(n4 + 255) / 256, 256, 0, stream>>>((float4*)xj, n4);

    physnet_atom_feat_kernel<<<(NA + 63) / 64, 128, 0, stream>>>(
        x, W_i, b_i, W_j, b_j, xi, xjf, NA);

    physnet_edge_kernel<<<(NE + 63) / 64, 128, 0, stream>>>(
        rbf, W_k2f, idx_i, idx_j, xjf, xj, NE);

    physnet_residual_out_kernel<<<(NA + 63) / 64, 128, 0, stream>>>(
        xi, xj, x, W_res1, b_res1, W_res2, b_res2, W_out, b_out, u, out, NA);
}
